// AdaptiveGatingNetwork_23957327577725
// MI455X (gfx1250) — matrix operand order repaired and checked by reading, not compiled, not run on hardware
//
#include <hip/hip_runtime.h>
#include <hip/hip_bf16.h>

#define V_ 512
#define D_ 128
#define H_ 64
#define E_ 4
#define L_ 60
#define B_ 4096
#define ADJ_ 184
#define FEAT_ 180
#define FEATP_ 192   // padded feature width (multiple of 64)
#define ADJP_ 192    // padded ADJ width (multiple of 64)

typedef __attribute__((ext_vector_type(16))) __bf16 v16bf;
typedef __attribute__((ext_vector_type(8)))  float  v8f;

union BFV {
    v16bf v;
    unsigned short u[16];
    uint4 q[2];
};

__device__ __forceinline__ unsigned short f2bf(float f) {
    unsigned int u = __float_as_uint(f);
    unsigned int r = u + 0x7FFFu + ((u >> 16) & 1u);  // round-to-nearest-even
    return (unsigned short)(r >> 16);
}

__device__ __forceinline__ float sigf(float x) { return 1.0f / (1.0f + expf(-x)); }
__device__ __forceinline__ int imax(int a, int b) { return a > b ? a : b; }
__device__ __forceinline__ int imin(int a, int b) { return a < b ? a : b; }

// Convert 16 consecutive (16B-aligned) floats to a bf16 WMMA chunk.
__device__ __forceinline__ v16bf cvt16_aligned(const float* p) {
    const float4* q = (const float4*)p;
    float4 a = q[0], b = q[1], c = q[2], d = q[3];
    BFV t;
    t.u[0]=f2bf(a.x); t.u[1]=f2bf(a.y); t.u[2]=f2bf(a.z); t.u[3]=f2bf(a.w);
    t.u[4]=f2bf(b.x); t.u[5]=f2bf(b.y); t.u[6]=f2bf(b.z); t.u[7]=f2bf(b.w);
    t.u[8]=f2bf(c.x); t.u[9]=f2bf(c.y); t.u[10]=f2bf(c.z); t.u[11]=f2bf(c.w);
    t.u[12]=f2bf(d.x); t.u[13]=f2bf(d.y); t.u[14]=f2bf(d.z); t.u[15]=f2bf(d.w);
    return t.v;
}

// ---------------------------------------------------------------------------
// emb (f32) -> bf16
// ---------------------------------------------------------------------------
__global__ __launch_bounds__(256) void convert_emb_kernel(const float* __restrict__ src,
                                                          unsigned short* __restrict__ dst,
                                                          int n) {
    int i = blockIdx.x * 256 + threadIdx.x;
    if (i < n) dst[i] = f2bf(src[i]);
}

// ---------------------------------------------------------------------------
// Pack a weight matrix W(N,K) f32 row-major into bf16 (Np,Kp), zero-padded.
// grid: (1, Np) with 256 threads; Kp <= 256.
// ---------------------------------------------------------------------------
__global__ __launch_bounds__(256) void pack_weight_kernel(const float* __restrict__ W,
                                                          unsigned short* __restrict__ out,
                                                          int N, int K, int Kp) {
    int n = blockIdx.y;
    int k = blockIdx.x * 256 + threadIdx.x;
    if (k < Kp)
        out[(size_t)n * Kp + k] =
            (n < N && k < K) ? f2bf(W[(size_t)n * K + k]) : (unsigned short)0;
}

// ---------------------------------------------------------------------------
// Fused bidirectional-LSTM direction kernel (see round-0 notes).
// Writes final h (bf16) into featsb[:, foff:foff+64], lda = FEATP_.
// ---------------------------------------------------------------------------
template <int REV>
__global__ __launch_bounds__(128) void lstm_kernel(const int* __restrict__ x,
                                                   const unsigned short* __restrict__ embbf,
                                                   const float* __restrict__ w_ih,   // (256,128)
                                                   const float* __restrict__ w_hh,   // (256,64)
                                                   const float* __restrict__ b_ih,
                                                   const float* __restrict__ b_hh,
                                                   unsigned short* __restrict__ featsb,
                                                   int foff) {
    __shared__ __align__(16) unsigned short hbuf[2][16 * 64];

    const int lane = threadIdx.x & 31;
    const int wv   = threadIdx.x >> 5;   // 0..3
    const int half = lane >> 4;          // 0/1
    const int mr   = lane & 15;          // row (A) / col (B,C)
    const int rb   = blockIdx.x * 16;    // batch-row base

    // resident B fragments + fused bias
    v16bf bx[4][4];
    v16bf bh[4][2];
    float bias[4];
#pragma unroll
    for (int q = 0; q < 4; ++q) {
        const int gcol = q * 64 + wv * 16 + mr;
        bias[q] = b_ih[gcol] + b_hh[gcol];
#pragma unroll
        for (int kt = 0; kt < 4; ++kt)
            bx[q][kt] = cvt16_aligned(w_ih + gcol * 128 + kt * 32 + half * 16);
#pragma unroll
        for (int kt = 0; kt < 2; ++kt)
            bh[q][kt] = cvt16_aligned(w_hh + gcol * 64 + kt * 32 + half * 16);
    }

    for (int i = threadIdx.x; i < 16 * 64; i += 128) hbuf[0][i] = 0;
    __syncthreads();

    float cst[8];
#pragma unroll
    for (int r = 0; r < 8; ++r) cst[r] = 0.0f;
    float hlast[8];

    int p = 0;
    for (int t = 0; t < L_; ++t) {
        const int tt  = REV ? (L_ - 1 - t) : t;
        const int tok = x[(rb + mr) * L_ + tt];

        v16bf ax[4];
        const uint4* erow = (const uint4*)(embbf + tok * 128);
#pragma unroll
        for (int kt = 0; kt < 4; ++kt) {
            BFV a;
            a.q[0] = erow[kt * 4 + half * 2];
            a.q[1] = erow[kt * 4 + half * 2 + 1];
            ax[kt] = a.v;
        }
        v16bf ah[2];
        const uint4* hrow = (const uint4*)(hbuf[p] + mr * 64);
#pragma unroll
        for (int kt = 0; kt < 2; ++kt) {
            BFV a;
            a.q[0] = hrow[kt * 4 + half * 2];
            a.q[1] = hrow[kt * 4 + half * 2 + 1];
            ah[kt] = a.v;
        }

        v8f g[4];
#pragma unroll
        for (int q = 0; q < 4; ++q) {
            v8f acc;
#pragma unroll
            for (int r = 0; r < 8; ++r) acc[r] = bias[q];
#pragma unroll
            for (int kt = 0; kt < 4; ++kt)
                acc = __builtin_amdgcn_wmma_f32_16x16x32_bf16(false, ax[kt], false,
                                                              bx[q][kt], (short)0, acc,
                                                              false, false);
#pragma unroll
            for (int kt = 0; kt < 2; ++kt)
                acc = __builtin_amdgcn_wmma_f32_16x16x32_bf16(false, ah[kt], false,
                                                              bh[q][kt], (short)0, acc,
                                                              false, false);
            g[q] = acc;
        }

        unsigned short* hw = hbuf[p ^ 1];
#pragma unroll
        for (int r = 0; r < 8; ++r) {
            float ig = sigf(g[0][r]);
            float fg = sigf(g[1][r]);
            float gg = tanhf(g[2][r]);
            float og = sigf(g[3][r]);
            float cn = fg * cst[r] + ig * gg;
            cst[r]   = cn;
            float hv = og * tanhf(cn);
            hlast[r] = hv;
            hw[(half * 8 + r) * 64 + wv * 16 + mr] = f2bf(hv);
        }
        __syncthreads();
        p ^= 1;
    }

#pragma unroll
    for (int r = 0; r < 8; ++r)
        featsb[(size_t)(rb + half * 8 + r) * FEATP_ + foff + wv * 16 + mr] = f2bf(hlast[r]);
}

// ---------------------------------------------------------------------------
// bf16-WMMA GEMM over pre-packed, zero-padded operands:
//   C(M,Np) = act( A(M,Kp)bf16 @ Wp(Np,Kp)bf16^T + bias )
// Block: 128 threads (4 waves); wave = one 16x16 tile; grid (M/16, Np/64).
// No masking in the K loop; padded W rows are zero, so padded output columns
// are exactly 0. Output is bf16 (padded buffer) or masked f32 (final logits).
// ---------------------------------------------------------------------------
__global__ __launch_bounds__(128) void gemm_bf16_kernel(
        const unsigned short* __restrict__ A, int lda,
        const unsigned short* __restrict__ Wp, int ldw,
        const float* __restrict__ bias, int nbias,
        void* __restrict__ Cout, int ldc,
        int Kp, int relu, int store_f32, int nstore) {
    const int lane = threadIdx.x & 31;
    const int wv   = threadIdx.x >> 5;
    const int half = lane >> 4;
    const int mr   = lane & 15;

    const int row0 = blockIdx.x * 16;
    const int ncol = blockIdx.y * 64 + wv * 16 + mr;

    v8f acc;
    float bv = (ncol < nbias) ? bias[ncol] : 0.0f;
#pragma unroll
    for (int r = 0; r < 8; ++r) acc[r] = bv;

    const unsigned short* arow = A + (size_t)(row0 + mr) * lda;
    const unsigned short* wrow = Wp + (size_t)ncol * ldw;

    for (int kb = 0; kb < Kp; kb += 32) {
        BFV a, w;
        a.q[0] = *(const uint4*)(arow + kb + half * 16);       // K [h*16, h*16+8)
        a.q[1] = *(const uint4*)(arow + kb + half * 16 + 8);   // K [h*16+8, h*16+16)
        w.q[0] = *(const uint4*)(wrow + kb + half * 16);       // K [h*16, h*16+8)
        w.q[1] = *(const uint4*)(wrow + kb + half * 16 + 8);   // K [h*16+8, h*16+16)
        acc = __builtin_amdgcn_wmma_f32_16x16x32_bf16(false, a.v, false, w.v,
                                                      (short)0, acc, false, false);
    }

    if (relu) {
#pragma unroll
        for (int r = 0; r < 8; ++r) acc[r] = fmaxf(acc[r], 0.0f);
    }

    if (store_f32) {
        if (ncol < nstore) {
            float* C = (float*)Cout;
#pragma unroll
            for (int r = 0; r < 8; ++r)
                C[(size_t)(row0 + half * 8 + r) * ldc + ncol] = acc[r];
        }
    } else {
        unsigned short* C = (unsigned short*)Cout;
#pragma unroll
        for (int r = 0; r < 8; ++r)
            C[(size_t)(row0 + half * 8 + r) * ldc + ncol] = f2bf(acc[r]);
    }
}

// ---------------------------------------------------------------------------
// Per-sequence statistics + tiny MLPs (e/l/c/p/x) -> featsb[:,128:180] (bf16),
// zeroes the pad columns [180:192). One thread per batch element.
// ---------------------------------------------------------------------------
__global__ __launch_bounds__(128) void stats_head_kernel(
        const int* __restrict__ x,
        const float* e_w1, const float* e_b1, const float* e_w2, const float* e_b2,
        const float* l_w1, const float* l_b1, const float* l_w2, const float* l_b2,
        const float* c_w1, const float* c_b1, const float* c_w2, const float* c_b2,
        const float* p_w1, const float* p_b1, const float* p_w2, const float* p_b2,
        const float* xw1,  const float* xb1,  const float* xw2,  const float* xb2,
        unsigned short* __restrict__ featsb) {
    int b = blockIdx.x * 128 + threadIdx.x;
    if (b >= B_) return;

    unsigned short cs[L_];
    unsigned char  cnt[V_];
    for (int i = 0; i < V_; ++i) cnt[i] = 0;

    int vlen = 0;
    for (int i = 0; i < L_; ++i) {
        int tk = x[(size_t)b * L_ + i];
        if (tk != 0) { cs[vlen++] = (unsigned short)tk; cnt[tk]++; }
    }
    const float vlenf = (float)imax(vlen, 1);

    float ent = 0.0f;
    int distinct = 0;
    for (int v = 0; v < V_; ++v) {
        if (cnt[v]) {
            ++distinct;
            float pb = (float)cnt[v] / vlenf;
            ent -= pb * logf(pb + 1e-8f);
        }
    }
    if (vlen <= 1) ent = 0.0f;

    int nrep = 0, ninc = 0, ndec = 0;
    for (int i = 0; i + 1 < vlen; ++i) {
        if (cs[i + 1] == cs[i]) ++nrep;
        else if (cs[i + 1] > cs[i]) ++ninc;
        else ++ndec;
    }
    const float den1 = (float)imax(vlen - 1, 1);
    float repeat = nrep / den1, inc = ninc / den1, dec = ndec / den1;

    int nper = 0;
    for (int i = 0; i + 2 < vlen; ++i)
        if (cs[i] == cs[i + 2]) ++nper;
    float per = (vlen >= 4) ? (float)nper / (float)imax(vlen - 2, 1) : 0.0f;

    float diversity = (float)distinct / vlenf;

    int maxrun = 0, run = 0;
    for (int i = 0; i < vlen; ++i) {
        run = (i > 0 && cs[i] == cs[i - 1]) ? run + 1 : 1;
        if (run > maxrun) maxrun = run;
    }
    float max_rep = (float)maxrun / vlenf;

    float pat[6] = {repeat, inc, dec, per, diversity, max_rep};
    if (vlen <= 1)
        for (int i = 0; i < 6; ++i) pat[i] = 0.0f;

    int ndist = 0;
    for (int i = 0; i + 1 < vlen; ++i) {
        int code = (int)cs[i] * V_ + (int)cs[i + 1];
        bool seen = false;
        for (int j = 0; j < i; ++j)
            if ((int)cs[j] * V_ + (int)cs[j + 1] == code) { seen = true; break; }
        if (!seen) ++ndist;
    }
    float comp_ratio   = (float)ndist / den1;
    float info_density = ent / logf((float)imax(vlen, 2));

    int wsel = imin(5, vlen / 2);
    float localc = 0.0f;
    if (wsel >= 2) {
        int w = wsel, nwin = vlen - w + 1;
        float su = 0.0f;
        for (int s = 0; s < nwin; ++s) {
            int u = 1;
            for (int k = 1; k < w; ++k) {
                bool nw = true;
                for (int m = 0; m < k; ++m)
                    if (cs[s + k] == cs[s + m]) { nw = false; break; }
                if (nw) ++u;
            }
            su += (float)u;
        }
        localc = su / (float)w / (float)imax(nwin, 1);
    }
    float trans = 1.0f - repeat;
    float cx[4] = {comp_ratio, info_density, localc, trans};
    if (vlen <= 1)
        for (int i = 0; i < 4; ++i) cx[i] = 0.0f;

    unsigned short* fr = featsb + (size_t)b * FEATP_;

    {   // e-MLP: 1 -> 16 -> 8  @ [128:136]
        float in = ent * 0.25f, h[16];
        for (int j = 0; j < 16; ++j) h[j] = fmaxf(e_w1[j] * in + e_b1[j], 0.0f);
        for (int o = 0; o < 8; ++o) {
            float s = e_b2[o];
            for (int j = 0; j < 16; ++j) s += e_w2[o * 16 + j] * h[j];
            fr[128 + o] = f2bf(s);
        }
    }
    {   // l-MLP: 1 -> 16 -> 8  @ [136:144]
        float in = (float)vlen / 60.0f, h[16];
        for (int j = 0; j < 16; ++j) h[j] = fmaxf(l_w1[j] * in + l_b1[j], 0.0f);
        for (int o = 0; o < 8; ++o) {
            float s = l_b2[o];
            for (int j = 0; j < 16; ++j) s += l_w2[o * 16 + j] * h[j];
            fr[136 + o] = f2bf(s);
        }
    }
    {   // c-MLP: 512 -> 32 -> 16  @ [144:160]  (sparse: <=60 nonzero bins)
        float h[32];
        for (int j = 0; j < 32; ++j) {
            float s = c_b1[j];
            const float* wr = c_w1 + (size_t)j * V_;
            for (int v = 0; v < V_; ++v)
                if (cnt[v]) s += wr[v] * ((float)cnt[v] / vlenf);
            h[j] = fmaxf(s, 0.0f);
        }
        for (int o = 0; o < 16; ++o) {
            float s = c_b2[o];
            for (int j = 0; j < 32; ++j) s += c_w2[o * 32 + j] * h[j];
            fr[144 + o] = f2bf(s);
        }
    }
    {   // p-MLP: 6 -> 24 -> 12  @ [160:172]
        float h[24];
        for (int j = 0; j < 24; ++j) {
            float s = p_b1[j];
            for (int i = 0; i < 6; ++i) s += p_w1[j * 6 + i] * pat[i];
            h[j] = fmaxf(s, 0.0f);
        }
        for (int o = 0; o < 12; ++o) {
            float s = p_b2[o];
            for (int j = 0; j < 24; ++j) s += p_w2[o * 24 + j] * h[j];
            fr[160 + o] = f2bf(s);
        }
    }
    {   // x-MLP: 4 -> 16 -> 8  @ [172:180]
        float h[16];
        for (int j = 0; j < 16; ++j) {
            float s = xb1[j];
            for (int i = 0; i < 4; ++i) s += xw1[j * 4 + i] * cx[i];
            h[j] = fmaxf(s, 0.0f);
        }
        for (int o = 0; o < 8; ++o) {
            float s = xb2[o];
            for (int j = 0; j < 16; ++j) s += xw2[o * 16 + j] * h[j];
            fr[172 + o] = f2bf(s);
        }
    }
    // zero the K padding so downstream WMMA reads contribute nothing
    for (int k = FEAT_; k < FEATP_; ++k) fr[k] = 0;
}

// ---------------------------------------------------------------------------
// softmax over E=4 logits, in-place on d_out
// ---------------------------------------------------------------------------
__global__ __launch_bounds__(256) void softmax4_kernel(float* __restrict__ out) {
    int b = blockIdx.x * 256 + threadIdx.x;
    if (b >= B_) return;
    float v0 = out[b * 4 + 0], v1 = out[b * 4 + 1], v2 = out[b * 4 + 2], v3 = out[b * 4 + 3];
    float m = fmaxf(fmaxf(v0, v1), fmaxf(v2, v3));
    float e0 = expf(v0 - m), e1 = expf(v1 - m), e2 = expf(v2 - m), e3 = expf(v3 - m);
    float inv = 1.0f / (e0 + e1 + e2 + e3);
    out[b * 4 + 0] = e0 * inv;
    out[b * 4 + 1] = e1 * inv;
    out[b * 4 + 2] = e2 * inv;
    out[b * 4 + 3] = e3 * inv;
}

// ---------------------------------------------------------------------------
extern "C" void kernel_launch(void* const* d_in, const int* in_sizes, int n_in,
                              void* d_out, int out_size, void* d_ws, size_t ws_size,
                              hipStream_t stream) {
    (void)in_sizes; (void)n_in; (void)out_size; (void)ws_size;

    const int*   x      = (const int*)d_in[0];
    const float* emb    = (const float*)d_in[1];
    const float* w_ih_f = (const float*)d_in[2];
    const float* w_hh_f = (const float*)d_in[3];
    const float* b_ih_f = (const float*)d_in[4];
    const float* b_hh_f = (const float*)d_in[5];
    const float* w_ih_r = (const float*)d_in[6];
    const float* w_hh_r = (const float*)d_in[7];
    const float* b_ih_r = (const float*)d_in[8];
    const float* b_hh_r = (const float*)d_in[9];
    const float* e_w1 = (const float*)d_in[10], *e_b1 = (const float*)d_in[11];
    const float* e_w2 = (const float*)d_in[12], *e_b2 = (const float*)d_in[13];
    const float* l_w1 = (const float*)d_in[14], *l_b1 = (const float*)d_in[15];
    const float* l_w2 = (const float*)d_in[16], *l_b2 = (const float*)d_in[17];
    const float* c_w1 = (const float*)d_in[18], *c_b1 = (const float*)d_in[19];
    const float* c_w2 = (const float*)d_in[20], *c_b2 = (const float*)d_in[21];
    const float* p_w1 = (const float*)d_in[22], *p_b1 = (const float*)d_in[23];
    const float* p_w2 = (const float*)d_in[24], *p_b2 = (const float*)d_in[25];
    const float* x_w1 = (const float*)d_in[26], *x_b1 = (const float*)d_in[27];
    const float* x_w2 = (const float*)d_in[28], *x_b2 = (const float*)d_in[29];
    const float* fp_w  = (const float*)d_in[30], *fp_b  = (const float*)d_in[31];
    const float* attn_w= (const float*)d_in[32], *attn_b= (const float*)d_in[33];
    const float* out_w = (const float*)d_in[34], *out_b = (const float*)d_in[35];
    const float* g_w1 = (const float*)d_in[36], *g_b1 = (const float*)d_in[37];
    const float* g_w2 = (const float*)d_in[38], *g_b2 = (const float*)d_in[39];
    const float* g_w3 = (const float*)d_in[40], *g_b3 = (const float*)d_in[41];
    const float* g_w4 = (const float*)d_in[42], *g_b4 = (const float*)d_in[43];

    // workspace carve-up (256B-aligned)
    char* base = (char*)d_ws;
    size_t off = 0;
    auto carve = [&](size_t bytes) -> void* {
        void* p = base + off;
        off += (bytes + 255) & ~(size_t)255;
        return p;
    };
    unsigned short* embbf  = (unsigned short*)carve((size_t)V_ * D_ * 2);
    unsigned short* featsb = (unsigned short*)carve((size_t)B_ * FEATP_ * 2);
    unsigned short* fb     = (unsigned short*)carve((size_t)B_ * ADJP_ * 2);
    unsigned short* vb     = (unsigned short*)carve((size_t)B_ * ADJP_ * 2);
    unsigned short* ab     = (unsigned short*)carve((size_t)B_ * ADJP_ * 2);
    unsigned short* h1b    = (unsigned short*)carve((size_t)B_ * 128 * 2);
    unsigned short* h2b    = (unsigned short*)carve((size_t)B_ * 64 * 2);
    unsigned short* h3b    = (unsigned short*)carve((size_t)B_ * 64 * 2);
    unsigned short* wp_fp  = (unsigned short*)carve((size_t)192 * 192 * 2);
    unsigned short* wp_v   = (unsigned short*)carve((size_t)192 * 192 * 2);
    unsigned short* wp_out = (unsigned short*)carve((size_t)192 * 192 * 2);
    unsigned short* wp_g1  = (unsigned short*)carve((size_t)128 * 192 * 2);
    unsigned short* wp_g2  = (unsigned short*)carve((size_t)64 * 128 * 2);
    unsigned short* wp_g3  = (unsigned short*)carve((size_t)64 * 64 * 2);
    unsigned short* wp_g4  = (unsigned short*)carve((size_t)64 * 32 * 2);
    float* logits = (float*)d_out;

    // 1) embedding -> bf16
    convert_emb_kernel<<<(V_ * D_) / 256, 256, 0, stream>>>(emb, embbf, V_ * D_);

    // 2) pack head weights to zero-padded bf16 (Np x Kp)
    pack_weight_kernel<<<dim3(1, 192), 256, 0, stream>>>(fp_w, wp_fp, ADJ_, FEAT_, 192);
    pack_weight_kernel<<<dim3(1, 192), 256, 0, stream>>>(attn_w + (size_t)2 * ADJ_ * ADJ_,
                                                         wp_v, ADJ_, ADJ_, 192);
    pack_weight_kernel<<<dim3(1, 192), 256, 0, stream>>>(out_w, wp_out, ADJ_, ADJ_, 192);
    pack_weight_kernel<<<dim3(1, 128), 256, 0, stream>>>(g_w1, wp_g1, 128, ADJ_, 192);
    pack_weight_kernel<<<dim3(1, 64),  256, 0, stream>>>(g_w2, wp_g2, 64, 128, 128);
    pack_weight_kernel<<<dim3(1, 64),  256, 0, stream>>>(g_w3, wp_g3, 32, 64, 64);
    pack_weight_kernel<<<dim3(1, 64),  256, 0, stream>>>(g_w4, wp_g4, E_, 32, 32);

    // 3) per-sequence statistics + small MLPs -> featsb[:,128:192]
    stats_head_kernel<<<B_ / 128, 128, 0, stream>>>(
        x,
        e_w1, e_b1, e_w2, e_b2,
        l_w1, l_b1, l_w2, l_b2,
        c_w1, c_b1, c_w2, c_b2,
        p_w1, p_b1, p_w2, p_b2,
        x_w1, x_b1, x_w2, x_b2,
        featsb);

    // 4) fused bidirectional LSTM (WMMA) -> featsb[:,0:64], featsb[:,64:128]
    lstm_kernel<0><<<B_ / 16, 128, 0, stream>>>(x, embbf, w_ih_f, w_hh_f, b_ih_f, b_hh_f, featsb, 0);
    lstm_kernel<1><<<B_ / 16, 128, 0, stream>>>(x, embbf, w_ih_r, w_hh_r, b_ih_r, b_hh_r, featsb, 64);

    const dim3 blk(128);
    // 5) f = feats @ fp_w^T + fp_b
    gemm_bf16_kernel<<<dim3(B_ / 16, 3), blk, 0, stream>>>(
        featsb, FEATP_, wp_fp, 192, fp_b, ADJ_, fb, ADJP_, 192, 0, 0, 0);
    // 6) v = f @ wv^T + bv
    gemm_bf16_kernel<<<dim3(B_ / 16, 3), blk, 0, stream>>>(
        fb, ADJP_, wp_v, 192, attn_b + 2 * ADJ_, ADJ_, vb, ADJP_, 192, 0, 0, 0);
    // 7) attended = v @ out_w^T + out_b
    gemm_bf16_kernel<<<dim3(B_ / 16, 3), blk, 0, stream>>>(
        vb, ADJP_, wp_out, 192, out_b, ADJ_, ab, ADJP_, 192, 0, 0, 0);
    // 8) gating head
    gemm_bf16_kernel<<<dim3(B_ / 16, 2), blk, 0, stream>>>(
        ab, ADJP_, wp_g1, 192, g_b1, 128, h1b, 128, 192, 1, 0, 0);
    gemm_bf16_kernel<<<dim3(B_ / 16, 1), blk, 0, stream>>>(
        h1b, 128, wp_g2, 128, g_b2, 64, h2b, 64, 128, 1, 0, 0);
    gemm_bf16_kernel<<<dim3(B_ / 16, 1), blk, 0, stream>>>(
        h2b, 64, wp_g3, 64, g_b3, 32, h3b, 64, 64, 1, 0, 0);
    gemm_bf16_kernel<<<dim3(B_ / 16, 1), blk, 0, stream>>>(
        h3b, 64, wp_g4, 32, g_b4, E_, logits, E_, 32, 0, 1, E_);

    // 9) softmax over E=4, in place on d_out
    softmax4_kernel<<<B_ / 256, 256, 0, stream>>>(logits);
}
